// Aggre_item_27814208209713
// MI455X (gfx1250) — matrix operand order repaired and checked by reading, not compiled, not run on hardware
//
#include <hip/hip_runtime.h>

// ---------------------------------------------------------------------------
// CDNA5 (gfx1250, wave32) implementation of the graph-attention aggregation.
// Edge pipeline (ln1 / att1 / att2) and node pipeline (ln2 / ln3) run on
// V_WMMA_F32_16X16X32_F16 with f16 operands staged through LDS; segment
// softmax uses an order-preserving u32 encoding + atomicMax, and the weighted
// segment-sum exploits sorted seg_ids with register accumulation + rare
// atomic flushes (sorted runs ~50 edges -> ~25x fewer atomics).
// ---------------------------------------------------------------------------

typedef _Float16 v16h __attribute__((ext_vector_type(16)));
typedef _Float16 v8h  __attribute__((ext_vector_type(8)));
typedef _Float16 v4h  __attribute__((ext_vector_type(4)));
typedef _Float16 v2h  __attribute__((ext_vector_type(2)));
typedef float    v8f  __attribute__((ext_vector_type(8)));

__device__ __forceinline__ v16h cat8(v8h a, v8h b) {
  return __builtin_shufflevector(a, b, 0,1,2,3,4,5,6,7,8,9,10,11,12,13,14,15);
}

__device__ __forceinline__ v8f wmma_f16(v16h a, v16h b, v8f c) {
  // 8 args: (neg_a, A, neg_b, B, c_mod, C, reuse_a, reuse_b)
  return __builtin_amdgcn_wmma_f32_16x16x32_f16(false, a, false, b, (short)0, c, false, false);
}

// A-fragment (16xK tile, row-major f16 in LDS, rowlen halves per row).
// lane<16: halves0-15 -> K=k0..k0+15; lane>=16: K=k0+16..k0+31 (matches bfrag).
__device__ __forceinline__ v16h afrag(const _Float16* t, int rowlen, int lane, int k0) {
  int hf = lane >> 4, m = lane & 15;
  const _Float16* p = t + m * rowlen + k0 + hf * 16;
  return cat8(*(const v8h*)p, *(const v8h*)(p + 8));
}

// B-fragment from weight W[n][k] (row-major, klen = K length). We need
// B[k][n] = W[n][k]. lane<16: col=n0+lane, K=k0..k0+15; lane>=16: col=n0+lane-16,
// K=k0+16..k0+31 (contiguous k order, same as afrag).
__device__ __forceinline__ v16h bfrag(const _Float16* w, int klen, int lane, int n0, int k0) {
  int hf = lane >> 4, n = n0 + (lane & 15);
  const _Float16* p = w + n * klen + k0 + hf * 16;
  return cat8(*(const v8h*)p, *(const v8h*)(p + 8));
}

__device__ __forceinline__ void st4h(_Float16* p, float4 v) {
  v4h h = { (_Float16)v.x, (_Float16)v.y, (_Float16)v.z, (_Float16)v.w };
  *(v4h*)p = h;
}

// Order-preserving float<->u32 encode for atomicMax on signed floats.
__device__ __forceinline__ unsigned fenc(float f) {
  unsigned u = __float_as_uint(f);
  return (u >> 31) ? ~u : (u | 0x80000000u);
}
__device__ __forceinline__ float fdec(unsigned u) {
  return __uint_as_float((u >> 31) ? (u & 0x7FFFFFFFu) : ~u);
}

// ---------------------------------------------------------------------------
__global__ __launch_bounds__(256) void init_ws(unsigned* __restrict__ smax,
                                               float* __restrict__ denom,
                                               float* __restrict__ num, int Bn) {
  int i = blockIdx.x * blockDim.x + threadIdx.x;
  if (i < Bn) { smax[i] = 0x007FFFFFu /* enc(-inf) */; denom[i] = 0.f; }
  if (i < Bn * 64) num[i] = 0.f;
}

// ---------------------------------------------------------------------------
// Kernel A: one wave per 16-edge tile. ln1 -> att1 -> att2 -> att3 score.
// ---------------------------------------------------------------------------
#define NWA 4
__global__ __launch_bounds__(128) void kA(
    const int* __restrict__ nodes, const int* __restrict__ item_ids,
    const int* __restrict__ rating_ids, const int* __restrict__ seg_ids,
    const float* __restrict__ user_emb, const float* __restrict__ item_emb,
    const float* __restrict__ rating_emb,
    const float* __restrict__ ln1_w, const float* __restrict__ ln1_b,
    const float* __restrict__ att1_w, const float* __restrict__ att1_b,
    const float* __restrict__ att2_w, const float* __restrict__ att2_b,
    const float* __restrict__ att3_w, const float* __restrict__ att3_b,
    _Float16* __restrict__ xg, float* __restrict__ sg, unsigned* __restrict__ smax) {
  __shared__ __align__(16) _Float16 LW1[64 * 128];
  __shared__ __align__(16) _Float16 LA1[64 * 128];
  __shared__ __align__(16) _Float16 LA2[64 * 64];
  __shared__ __align__(16) _Float16 cIR[NWA][16 * 128];
  __shared__ __align__(16) _Float16 cUu[NWA][16 * 64];
  __shared__ __align__(16) _Float16 cX [NWA][16 * 64];
  __shared__ __align__(16) _Float16 cA [NWA][16 * 64];

  for (int i = threadIdx.x; i < 64 * 128; i += blockDim.x) LW1[i] = (_Float16)ln1_w[i];
  for (int i = threadIdx.x; i < 64 * 128; i += blockDim.x) LA1[i] = (_Float16)att1_w[i];
  for (int i = threadIdx.x; i < 64 * 64;  i += blockDim.x) LA2[i] = (_Float16)att2_w[i];
  __syncthreads();

  const int wv = threadIdx.x >> 5, lane = threadIdx.x & 31;
  const long e0 = ((long)blockIdx.x * NWA + wv) * 16;
  _Float16* IR = cIR[wv]; _Float16* U = cUu[wv]; _Float16* X = cX[wv]; _Float16* A = cA[wv];

  // Gather 16 edges: concat(item,rating) -> IR[16][128], user[seg] -> U[16][64].
  {
    int m = lane >> 1, hh = lane & 1;
    long e = e0 + m;
    int it = item_ids[e], rt = rating_ids[e], ur = nodes[seg_ids[e]];
    const float* ip = item_emb   + (long)it * 64 + hh * 32;
    const float* rp = rating_emb + (long)rt * 64 + hh * 32;
    const float* up = user_emb   + (long)ur * 64 + hh * 32;
#pragma unroll
    for (int j = 0; j < 32; j += 4) {
      st4h(&IR[m * 128 + hh * 32 + j],      *(const float4*)(ip + j));
      st4h(&IR[m * 128 + 64 + hh * 32 + j], *(const float4*)(rp + j));
      st4h(&U [m * 64  + hh * 32 + j],      *(const float4*)(up + j));
    }
  }

  const int hf = lane >> 4, nl = lane & 15;

  // x = relu(IR @ ln1_w^T + b1): 4 N-tiles x 4 K-steps of wmma.
#pragma unroll
  for (int nt = 0; nt < 4; ++nt) {
    v8f c = {};
#pragma unroll
    for (int kt = 0; kt < 4; ++kt)
      c = wmma_f16(afrag(IR, 128, lane, kt * 32), bfrag(LW1, 128, lane, nt * 16, kt * 32), c);
    float bn = ln1_b[nt * 16 + nl];
#pragma unroll
    for (int r = 0; r < 8; ++r)
      X[(r + hf * 8) * 64 + nt * 16 + nl] = (_Float16)fmaxf(c[r] + bn, 0.f);
  }

  // Spill x tile (f16) to global for the softmax-accumulate pass.
  {
    int m = lane & 15, hh = lane >> 4;
    _Float16* gp = xg + (e0 + m) * 64 + hh * 32;
    const _Float16* lp = &X[m * 64 + hh * 32];
#pragma unroll
    for (int j = 0; j < 32; j += 8) *(v8h*)(gp + j) = *(const v8h*)(lp + j);
  }

  // a1 = relu(cat(x, user) @ att1_w^T + b): K = 64(x) + 64(user).
#pragma unroll
  for (int nt = 0; nt < 4; ++nt) {
    v8f c = {};
    c = wmma_f16(afrag(X, 64, lane, 0),  bfrag(LA1, 128, lane, nt * 16, 0),  c);
    c = wmma_f16(afrag(X, 64, lane, 32), bfrag(LA1, 128, lane, nt * 16, 32), c);
    c = wmma_f16(afrag(U, 64, lane, 0),  bfrag(LA1, 128, lane, nt * 16, 64), c);
    c = wmma_f16(afrag(U, 64, lane, 32), bfrag(LA1, 128, lane, nt * 16, 96), c);
    float bn = att1_b[nt * 16 + nl];
#pragma unroll
    for (int r = 0; r < 8; ++r)
      A[(r + hf * 8) * 64 + nt * 16 + nl] = (_Float16)fmaxf(c[r] + bn, 0.f);
  }

  // a2 = relu(a1 @ att2_w^T + b); s = a2 . att3_w + b, reduced per row.
  float psum[8] = {0, 0, 0, 0, 0, 0, 0, 0};
#pragma unroll
  for (int nt = 0; nt < 4; ++nt) {
    v8f c = {};
    c = wmma_f16(afrag(A, 64, lane, 0),  bfrag(LA2, 64, lane, nt * 16, 0),  c);
    c = wmma_f16(afrag(A, 64, lane, 32), bfrag(LA2, 64, lane, nt * 16, 32), c);
    float bn = att2_b[nt * 16 + nl];
    float w3 = att3_w[nt * 16 + nl];
#pragma unroll
    for (int r = 0; r < 8; ++r) psum[r] += fmaxf(c[r] + bn, 0.f) * w3;
  }
  // Reduce across the 16 lanes of each half-wave (columns of the D tile).
#pragma unroll
  for (int off = 1; off < 16; off <<= 1)
#pragma unroll
    for (int r = 0; r < 8; ++r) psum[r] += __shfl_xor(psum[r], off, 32);

  float b3 = att3_b[0];
  if (nl < 8) {  // lanes 0-7 hold rows 0-7, lanes 16-23 hold rows 8-15
    int r = nl, row = r + hf * 8;
    float sv = psum[r] + b3;
    sg[e0 + row] = sv;
    atomicMax(&smax[seg_ids[e0 + row]], fenc(sv));
  }
}

// ---------------------------------------------------------------------------
// Kernel B: one wave per 64-edge chunk. seg_ids is sorted, so accumulate
// e*x and sum(e) in registers across each segment run and flush with atomics
// only at run boundaries (expected ~2.3 runs per chunk vs 64 edges).
// Lane l owns feature columns 2l and 2l+1.
// ---------------------------------------------------------------------------
#define KB_CHUNK 64
__global__ __launch_bounds__(256) void kB(
    const int* __restrict__ seg_ids, const float* __restrict__ sg,
    const _Float16* __restrict__ xg, const unsigned* __restrict__ smax,
    float* __restrict__ denom, float* __restrict__ num) {
  const int lane = threadIdx.x & 31;
  const long c0 = ((long)blockIdx.x * (blockDim.x >> 5) + (threadIdx.x >> 5)) * KB_CHUNK;
  const int d = lane * 2;

  float acc0 = 0.f, acc1 = 0.f, dsum = 0.f;
  int curseg = __builtin_amdgcn_readfirstlane(seg_ids[c0]);

  for (int i = 0; i < KB_CHUNK; ++i) {
    long e = c0 + i;
    int sgid = __builtin_amdgcn_readfirstlane(seg_ids[e]);
    if (sgid != curseg) {  // wave-uniform scalar branch
      atomicAdd(&num[(long)curseg * 64 + d],     acc0);
      atomicAdd(&num[(long)curseg * 64 + d + 1], acc1);
      if (lane == 0) atomicAdd(&denom[curseg], dsum);
      acc0 = 0.f; acc1 = 0.f; dsum = 0.f;
      curseg = sgid;
    }
    float ev = expf(sg[e] - fdec(smax[sgid]));
    v2h xv = *(const v2h*)(xg + e * 64 + d);
    acc0 += ev * (float)xv.x;
    acc1 += ev * (float)xv.y;
    dsum += ev;
  }
  atomicAdd(&num[(long)curseg * 64 + d],     acc0);
  atomicAdd(&num[(long)curseg * 64 + d + 1], acc1);
  if (lane == 0) atomicAdd(&denom[curseg], dsum);
}

// ---------------------------------------------------------------------------
// Kernel C: one wave per 16-node tile. hI = relu((num/denom)@ln2^T + b2),
// feats = relu(cat(user, hI)@ln3^T + b3).
// ---------------------------------------------------------------------------
#define NWC 4
__global__ __launch_bounds__(128) void kC(
    const int* __restrict__ nodes, const float* __restrict__ user_emb,
    const float* __restrict__ num, const float* __restrict__ denom,
    const float* __restrict__ ln2_w, const float* __restrict__ ln2_b,
    const float* __restrict__ ln3_w, const float* __restrict__ ln3_b,
    float* __restrict__ outp) {
  __shared__ __align__(16) _Float16 LW2[64 * 64];
  __shared__ __align__(16) _Float16 LW3[64 * 128];
  __shared__ __align__(16) _Float16 cH [NWC][16 * 64];
  __shared__ __align__(16) _Float16 cUu[NWC][16 * 64];
  __shared__ __align__(16) _Float16 cHI[NWC][16 * 64];

  for (int i = threadIdx.x; i < 64 * 64;  i += blockDim.x) LW2[i] = (_Float16)ln2_w[i];
  for (int i = threadIdx.x; i < 64 * 128; i += blockDim.x) LW3[i] = (_Float16)ln3_w[i];
  __syncthreads();

  const int wv = threadIdx.x >> 5, lane = threadIdx.x & 31;
  const int n0 = (blockIdx.x * NWC + wv) * 16;
  _Float16* H = cH[wv]; _Float16* U = cUu[wv]; _Float16* HI = cHI[wv];

  {
    int m = lane >> 1, hh = lane & 1;
    int node = n0 + m;
    int ur = nodes[node];
    float inv = 1.f / fmaxf(denom[node], 1e-12f);
    const float* np = num + (long)node * 64 + hh * 32;
    const float* up = user_emb + (long)ur * 64 + hh * 32;
#pragma unroll
    for (int j = 0; j < 32; j += 4) {
      float4 v = *(const float4*)(np + j);
      v.x *= inv; v.y *= inv; v.z *= inv; v.w *= inv;
      st4h(&H[m * 64 + hh * 32 + j], v);
      st4h(&U[m * 64 + hh * 32 + j], *(const float4*)(up + j));
    }
  }

  const int hf = lane >> 4, nl = lane & 15;

#pragma unroll
  for (int nt = 0; nt < 4; ++nt) {
    v8f c = {};
    c = wmma_f16(afrag(H, 64, lane, 0),  bfrag(LW2, 64, lane, nt * 16, 0),  c);
    c = wmma_f16(afrag(H, 64, lane, 32), bfrag(LW2, 64, lane, nt * 16, 32), c);
    float bn = ln2_b[nt * 16 + nl];
#pragma unroll
    for (int r = 0; r < 8; ++r)
      HI[(r + hf * 8) * 64 + nt * 16 + nl] = (_Float16)fmaxf(c[r] + bn, 0.f);
  }

#pragma unroll
  for (int nt = 0; nt < 4; ++nt) {
    v8f c = {};
    c = wmma_f16(afrag(U, 64, lane, 0),   bfrag(LW3, 128, lane, nt * 16, 0),  c);
    c = wmma_f16(afrag(U, 64, lane, 32),  bfrag(LW3, 128, lane, nt * 16, 32), c);
    c = wmma_f16(afrag(HI, 64, lane, 0),  bfrag(LW3, 128, lane, nt * 16, 64), c);
    c = wmma_f16(afrag(HI, 64, lane, 32), bfrag(LW3, 128, lane, nt * 16, 96), c);
    float bn = ln3_b[nt * 16 + nl];
#pragma unroll
    for (int r = 0; r < 8; ++r)
      outp[(long)(n0 + r + hf * 8) * 64 + nt * 16 + nl] = fmaxf(c[r] + bn, 0.f);
  }
}

// ---------------------------------------------------------------------------
extern "C" void kernel_launch(void* const* d_in, const int* in_sizes, int n_in,
                              void* d_out, int out_size, void* d_ws, size_t ws_size,
                              hipStream_t stream) {
  const int*   nodes      = (const int*)d_in[0];
  const int*   item_ids   = (const int*)d_in[1];
  const int*   rating_ids = (const int*)d_in[2];
  const int*   seg_ids    = (const int*)d_in[3];
  const float* user_emb   = (const float*)d_in[4];
  const float* item_emb   = (const float*)d_in[5];
  const float* rating_emb = (const float*)d_in[6];
  const float* ln1_w  = (const float*)d_in[7];
  const float* ln1_b  = (const float*)d_in[8];
  const float* ln2_w  = (const float*)d_in[9];
  const float* ln2_b  = (const float*)d_in[10];
  const float* ln3_w  = (const float*)d_in[11];
  const float* ln3_b  = (const float*)d_in[12];
  const float* att1_w = (const float*)d_in[13];
  const float* att1_b = (const float*)d_in[14];
  const float* att2_w = (const float*)d_in[15];
  const float* att2_b = (const float*)d_in[16];
  const float* att3_w = (const float*)d_in[17];
  const float* att3_b = (const float*)d_in[18];
  float* outp = (float*)d_out;

  const int E  = in_sizes[1];   // 819200
  const int Bn = in_sizes[0];   // 16384

  // Workspace layout: x(f16 E*64) | s(f32 E) | smax(u32 B) | denom(f32 B) | num(f32 B*64)
  char* ws = (char*)d_ws;
  _Float16* xg = (_Float16*)ws;
  size_t off = (size_t)E * 64 * sizeof(_Float16);
  float*    sg    = (float*)(ws + off);    off += (size_t)E * sizeof(float);
  unsigned* smax  = (unsigned*)(ws + off); off += (size_t)Bn * sizeof(unsigned);
  float*    denom = (float*)(ws + off);    off += (size_t)Bn * sizeof(float);
  float*    num   = (float*)(ws + off);

  init_ws<<<(Bn * 64 + 255) / 256, 256, 0, stream>>>(smax, denom, num, Bn);

  kA<<<(E / 16) / NWA, 128, 0, stream>>>(
      nodes, item_ids, rating_ids, seg_ids, user_emb, item_emb, rating_emb,
      ln1_w, ln1_b, att1_w, att1_b, att2_w, att2_b, att3_w, att3_b,
      xg, sg, smax);

  kB<<<(E / KB_CHUNK) / 8, 256, 0, stream>>>(seg_ids, sg, xg, smax, denom, num);

  kC<<<(Bn / 16) / NWC, 128, 0, stream>>>(
      nodes, user_emb, num, denom, ln2_w, ln2_b, ln3_w, ln3_b, outp);
}
